// Attention_25262997635344
// MI455X (gfx1250) — matrix operand order repaired and checked by reading, not compiled, not run on hardware
//
#include <hip/hip_runtime.h>
#include <hip/hip_bf16.h>
#include <math.h>

// ---------------------------------------------------------------------------
// Types
// ---------------------------------------------------------------------------
typedef __bf16 bf16;
typedef __attribute__((ext_vector_type(16))) __bf16 v16bf;
typedef __attribute__((ext_vector_type(8)))  float  v8f;
typedef __attribute__((ext_vector_type(4)))  unsigned int v4u;

union AFrag {
    v16bf v;
    v4u   u[2];
    bf16  e[16];
};

__device__ inline bf16 to_bf16(float f) { return (bf16)f; }

// LDS B-subtile row stride in elements: 32 data + 8 pad -> 80B rows.
// Keeps 16B alignment for b128 reads and makes the 16-lane fragment
// read pattern hit 16 distinct bank groups (stride 20 banks mod 64).
#define BSTRIDE 40
#define BSUB    (128 * BSTRIDE)   // one 128(N) x 32(K) subtile, in elements

// ---------------------------------------------------------------------------
// WMMA fragment loads (layouts per CDNA5 ISA 7.12.2, wave32)
//
// A 16x32 bf16 (row-major source, leading dim lda):
//   lane l (l<16):  row = l,     K = {0..7, 16..23}
//   lane l (l>=16): row = l-16,  K = {8..15, 24..31}
// B 32x16 bf16 from Bt stored [N][K] row-major (leading dim ldb):
//   lane n (n<16):  col = n,    K = {0..7, 16..23}
//   lane n (n>=16): col = n-16, K = {8..15, 24..31}
// ---------------------------------------------------------------------------
__device__ inline v16bf load_A16x32(const bf16* __restrict__ A, int lda,
                                    int row0, int k0) {
    const int lane = threadIdx.x & 31;
    const int half = lane >> 4;
    const int m    = lane & 15;
    const bf16* p = A + (size_t)(row0 + m) * lda + k0 + half * 8;
    AFrag f;
    f.u[0] = *(const v4u*)(p);        // K +0..7   (half0) / +8..15  (half1)
    f.u[1] = *(const v4u*)(p + 16);   // K +16..23 (half0) / +24..31 (half1)
    return f.v;
}

__device__ inline v16bf load_B32x16(const bf16* __restrict__ Bt, int ldb,
                                    int n0, int k0) {
    const int lane = threadIdx.x & 31;
    const int half = lane >> 4;
    const int n    = lane & 15;
    const bf16* p = Bt + (size_t)(n0 + n) * ldb + k0 + half * 8;
    AFrag f;
    f.u[0] = *(const v4u*)(p);
    f.u[1] = *(const v4u*)(p + 16);
    return f.v;
}

// B fragment from a padded LDS subtile (cols relative to tile).
__device__ inline v16bf load_B_lds(const bf16* base, int n0) {
    const int lane = threadIdx.x & 31;
    const int half = lane >> 4;
    const int n    = lane & 15;
    const bf16* p = base + (n0 + n) * BSTRIDE + half * 8;
    AFrag f;
    f.u[0] = *(const v4u*)(p);
    f.u[1] = *(const v4u*)(p + 16);
    return f.v;
}

__device__ inline v8f wmma_bf16(v16bf a, v16bf b, v8f c) {
    return __builtin_amdgcn_wmma_f32_16x16x32_bf16(
        false, a, false, b, (short)0, c, false, false);
}

// ---------------------------------------------------------------------------
// Async global->LDS staging of a 128(N) x 64(K) bf16 tile (16 KB payload),
// stored as two 32-K subtiles with padded rows. 1024 x 16B chunks,
// 4 per thread (256 threads). ASYNCcnt-tracked.
// ---------------------------------------------------------------------------
__device__ inline void stage_Btile64(const bf16* __restrict__ gbase, // row 0 of tile
                                     int ldk, int k0,
                                     bf16* lds_base, int tid) {
    int c = tid * 4;
    #pragma unroll
    for (int i = 0; i < 4; ++i, ++c) {
        const int rown = c >> 3;            // 0..127
        const int sub  = (c >> 2) & 1;      // which 32-K subtile
        const int koff = (c & 3) * 8;       // element offset within subtile row
        const bf16* g = gbase + (size_t)rown * ldk + k0 + sub * 32 + koff;
        const unsigned l =
            (unsigned)(size_t)(lds_base + sub * BSUB + rown * BSTRIDE + koff);
        asm volatile("global_load_async_to_lds_b128 %0, %1, off"
                     :: "v"(l), "v"(g) : "memory");
    }
}

// 16-lane butterfly reductions (C-tile rows live across lanes 0..15 / 16..31).
__device__ inline float rowmax16(float v) {
    #pragma unroll
    for (int m = 8; m >= 1; m >>= 1) v = fmaxf(v, __shfl_xor(v, m, 16));
    return v;
}
__device__ inline float rowsum16(float v) {
    #pragma unroll
    for (int m = 8; m >= 1; m >>= 1) v += __shfl_xor(v, m, 16);
    return v;
}

// ---------------------------------------------------------------------------
// Conversion kernels
// ---------------------------------------------------------------------------
__global__ __launch_bounds__(256) void cvt_f32_bf16(const float* __restrict__ in,
                                                    bf16* __restrict__ out, int n) {
    int i = blockIdx.x * 256 + threadIdx.x;
    if (i < n) out[i] = to_bf16(in[i]);
}

// in: [K][N] f32 row-major  ->  out: [N][K] bf16 row-major
__global__ __launch_bounds__(256) void transpose_cvt(const float* __restrict__ in,
                                                     bf16* __restrict__ out,
                                                     int K, int N) {
    int i = blockIdx.x * 256 + threadIdx.x;
    if (i < K * N) {
        int k = i / N, n = i - k * N;
        out[(size_t)n * K + k] = to_bf16(in[i]);
    }
}

// ---------------------------------------------------------------------------
// QKV GEMM: qkv[4096,3072] = X[4096,1024] @ W[1024,3072] + b
// Block = 256 thr (8 waves), block tile 128(M) x 128(N), K step 64.
// Waves: 4(M) x 2(N); wave tile 32 x 64 (8 accumulators, B reused for 2 rows).
// B tile double-buffered in LDS via global_load_async_to_lds_b128
// (16 WMMAs per barrier pair).
// Epilogue splits into q/k ([B,S,H*D] bf16) and v transposed ([B,H,D,S] bf16).
// ---------------------------------------------------------------------------
__global__ __launch_bounds__(256) void gemm_qkv(
    const bf16* __restrict__ X,      // [4096,1024]
    const bf16* __restrict__ Wt,     // [3072,1024]  (transposed weights)
    const float* __restrict__ bias,  // [3072]
    bf16* __restrict__ qb,           // [4096,1024]
    bf16* __restrict__ kb,           // [4096,1024]
    bf16* __restrict__ vtb) {        // [B=4][H=16][D=64][S=1024]
    __shared__ bf16 Bsh[2][2 * BSUB];   // [buf][subtile]

    const int tid  = threadIdx.x;
    const int wave = tid >> 5;
    const int mw = wave & 3, nw = wave >> 2;
    const int row0  = blockIdx.x * 128 + mw * 32;
    const int ncol0 = blockIdx.y * 128;          // block N base
    const bf16* Wtile = Wt + (size_t)ncol0 * 1024;

    v8f acc[8];
    #pragma unroll
    for (int t = 0; t < 8; ++t) acc[t] = 0.0f;

    stage_Btile64(Wtile, 1024, 0, &Bsh[0][0], tid);

    for (int k0 = 0; k0 < 1024; k0 += 64) {
        const int buf = (k0 >> 6) & 1;
        __syncthreads();  // all reads of buf^1 (previous iter) finished
        if (k0 + 64 < 1024) {
            stage_Btile64(Wtile, 1024, k0 + 64, &Bsh[buf ^ 1][0], tid);
            asm volatile("s_wait_asynccnt 4" ::: "memory");  // current buf landed
        } else {
            asm volatile("s_wait_asynccnt 0" ::: "memory");
        }
        __syncthreads();  // every wave's chunks of buf are visible

        if (k0 + 128 < 1024)
            __builtin_prefetch(X + (size_t)row0 * 1024 + k0 + 128, 0, 3);
        #pragma unroll
        for (int kh = 0; kh < 2; ++kh) {
            v16bf a0 = load_A16x32(X, 1024, row0,      k0 + kh * 32);
            v16bf a1 = load_A16x32(X, 1024, row0 + 16, k0 + kh * 32);
            #pragma unroll
            for (int t = 0; t < 4; ++t) {
                v16bf b = load_B_lds(&Bsh[buf][kh * BSUB], nw * 64 + t * 16);
                acc[t]     = wmma_bf16(a0, b, acc[t]);
                acc[4 + t] = wmma_bf16(a1, b, acc[4 + t]);
            }
        }
    }

    const int lane = tid & 31;
    const int g = lane >> 4, n = lane & 15;
    #pragma unroll
    for (int s = 0; s < 2; ++s) {          // M subtiles (+0 / +16)
        #pragma unroll
        for (int t = 0; t < 4; ++t) {
            const int ncol = ncol0 + nw * 64 + t * 16 + n;
            const float bs = bias[ncol];
            #pragma unroll
            for (int r = 0; r < 8; ++r) {
                const int row = row0 + s * 16 + r + 8 * g;   // global token
                const float val = acc[s * 4 + t][r] + bs;
                if (ncol < 1024) {
                    qb[(size_t)row * 1024 + ncol] = to_bf16(val);
                } else if (ncol < 2048) {
                    kb[(size_t)row * 1024 + (ncol - 1024)] = to_bf16(val);
                } else {
                    const int c = ncol - 2048;
                    const int h = c >> 6, d = c & 63;
                    const int b = row >> 10, tok = row & 1023;
                    vtb[(((size_t)(b * 16 + h)) * 64 + d) * 1024 + tok] = to_bf16(val);
                }
            }
        }
    }
}

// ---------------------------------------------------------------------------
// Flash attention: one wave per (b, h, 16-row query block).
// Online softmax over 32-key chunks; both matmuls via WMMA bf16.
// ---------------------------------------------------------------------------
__global__ __launch_bounds__(32) void attn_kernel(
    const bf16* __restrict__ qg,   // [B,S,H*D]
    const bf16* __restrict__ kg,   // [B,S,H*D]
    const bf16* __restrict__ vtg,  // [B,H,D,S]
    bf16* __restrict__ ag) {       // [B,S,H*D]
    __shared__ bf16 P_lds[16 * 32];

    const int mblk = blockIdx.x & 63;
    const int bh   = blockIdx.x >> 6;
    const int b    = bh >> 4, h = bh & 15;
    const int tok0 = mblk * 16;

    const bf16* Qb = qg  + (size_t)b * 1024 * 1024 + h * 64;
    const bf16* Kb = kg  + (size_t)b * 1024 * 1024 + h * 64;
    const bf16* Vt = vtg + (size_t)(b * 16 + h) * 64 * 1024;

    const int lane = threadIdx.x & 31;
    const int g = lane >> 4, n = lane & 15;

    // Q fragments (16 rows x 64 d, two K=32 chunks), held in registers.
    const v16bf qa0 = load_A16x32(Qb, 1024, tok0, 0);
    const v16bf qa1 = load_A16x32(Qb, 1024, tok0, 32);

    float m_i[8], l_i[8];
    v8f o[4];
    #pragma unroll
    for (int r = 0; r < 8; ++r) { m_i[r] = -1e30f; l_i[r] = 0.0f; }
    #pragma unroll
    for (int t = 0; t < 4; ++t) o[t] = 0.0f;

    const float scale = 0.125f;  // 1/sqrt(64)
    const int nchunks = (tok0 + 16 + 31) >> 5;

    for (int jc = 0; jc < nchunks; ++jc) {
        const int j0 = jc * 32;

        // S tiles: keys j0..j0+15 and j0+16..j0+31
        v8f s0 = 0.0f, s1 = 0.0f;
        s0 = wmma_bf16(qa0, load_B32x16(Kb, 1024, j0,      0), s0);
        s0 = wmma_bf16(qa1, load_B32x16(Kb, 1024, j0,     32), s0);
        s1 = wmma_bf16(qa0, load_B32x16(Kb, 1024, j0 + 16,  0), s1);
        s1 = wmma_bf16(qa1, load_B32x16(Kb, 1024, j0 + 16, 32), s1);

        // scale + causal mask + online softmax
        float p0[8], p1[8];
        #pragma unroll
        for (int r = 0; r < 8; ++r) {
            const int row = tok0 + r + 8 * g;
            float a0 = s0[r] * scale;
            float a1 = s1[r] * scale;
            if (j0 + n      > row) a0 = -1e30f;
            if (j0 + 16 + n > row) a1 = -1e30f;

            const float mloc = rowmax16(fmaxf(a0, a1));
            const float mnew = fmaxf(m_i[r], mloc);
            const float alpha = __expf(m_i[r] - mnew);
            a0 = __expf(a0 - mnew);
            a1 = __expf(a1 - mnew);
            l_i[r] = l_i[r] * alpha + rowsum16(a0 + a1);
            m_i[r] = mnew;
            p0[r] = a0; p1[r] = a1;
            #pragma unroll
            for (int t = 0; t < 4; ++t) o[t][r] *= alpha;
        }

        // Relay P (16 rows x 32 keys) into A-fragment layout via LDS.
        #pragma unroll
        for (int r = 0; r < 8; ++r) {
            const int row = r + 8 * g;
            P_lds[row * 32 + n]      = to_bf16(p0[r]);
            P_lds[row * 32 + 16 + n] = to_bf16(p1[r]);
        }
        __syncthreads();  // single-wave WG: ordering fence only

        AFrag pf;
        {
            const int m = lane & 15;
            const int kb8 = (lane >> 4) * 8;
            #pragma unroll
            for (int i = 0; i < 8; ++i) {
                pf.e[i]     = P_lds[m * 32 + kb8 + i];       // K {0..7} / {8..15}
                pf.e[8 + i] = P_lds[m * 32 + 16 + kb8 + i];  // K {16..23} / {24..31}
            }
        }
        __syncthreads();

        // O += P @ V  (B operand from transposed V: contiguous keys per d-row)
        #pragma unroll
        for (int t = 0; t < 4; ++t)
            o[t] = wmma_bf16(pf.v, load_B32x16(Vt, 1024, t * 16, j0), o[t]);
    }

    // Normalize and write merged-head output [B,S,H*D] in bf16.
    #pragma unroll
    for (int r = 0; r < 8; ++r) {
        const int row = tok0 + r + 8 * g;
        const float inv = 1.0f / l_i[r];
        #pragma unroll
        for (int t = 0; t < 4; ++t) {
            const int col = h * 64 + t * 16 + n;
            ag[((size_t)b * 1024 + row) * 1024 + col] = to_bf16(o[t][r] * inv);
        }
    }
}

// ---------------------------------------------------------------------------
// Proj GEMM: out f32 [4096,1024] = A_bf16 @ Wp + bias
// Same structure as gemm_qkv (128x128 block, K step 64, async B tile).
// ---------------------------------------------------------------------------
__global__ __launch_bounds__(256) void gemm_proj(
    const bf16* __restrict__ A,      // [4096,1024]
    const bf16* __restrict__ Wt,     // [1024,1024] transposed
    const float* __restrict__ bias,  // [1024]
    float* __restrict__ out) {       // [4096,1024]
    __shared__ bf16 Bsh[2][2 * BSUB];

    const int tid  = threadIdx.x;
    const int wave = tid >> 5;
    const int mw = wave & 3, nw = wave >> 2;
    const int row0  = blockIdx.x * 128 + mw * 32;
    const int ncol0 = blockIdx.y * 128;
    const bf16* Wtile = Wt + (size_t)ncol0 * 1024;

    v8f acc[8];
    #pragma unroll
    for (int t = 0; t < 8; ++t) acc[t] = 0.0f;

    stage_Btile64(Wtile, 1024, 0, &Bsh[0][0], tid);

    for (int k0 = 0; k0 < 1024; k0 += 64) {
        const int buf = (k0 >> 6) & 1;
        __syncthreads();
        if (k0 + 64 < 1024) {
            stage_Btile64(Wtile, 1024, k0 + 64, &Bsh[buf ^ 1][0], tid);
            asm volatile("s_wait_asynccnt 4" ::: "memory");
        } else {
            asm volatile("s_wait_asynccnt 0" ::: "memory");
        }
        __syncthreads();

        if (k0 + 128 < 1024)
            __builtin_prefetch(A + (size_t)row0 * 1024 + k0 + 128, 0, 3);
        #pragma unroll
        for (int kh = 0; kh < 2; ++kh) {
            v16bf a0 = load_A16x32(A, 1024, row0,      k0 + kh * 32);
            v16bf a1 = load_A16x32(A, 1024, row0 + 16, k0 + kh * 32);
            #pragma unroll
            for (int t = 0; t < 4; ++t) {
                v16bf b = load_B_lds(&Bsh[buf][kh * BSUB], nw * 64 + t * 16);
                acc[t]     = wmma_bf16(a0, b, acc[t]);
                acc[4 + t] = wmma_bf16(a1, b, acc[4 + t]);
            }
        }
    }

    const int lane = tid & 31;
    const int g = lane >> 4, n = lane & 15;
    #pragma unroll
    for (int s = 0; s < 2; ++s) {
        #pragma unroll
        for (int t = 0; t < 4; ++t) {
            const int ncol = ncol0 + nw * 64 + t * 16 + n;
            const float bs = bias[ncol];
            #pragma unroll
            for (int r = 0; r < 8; ++r) {
                const int row = row0 + s * 16 + r + 8 * g;
                out[(size_t)row * 1024 + ncol] = acc[s * 4 + t][r] + bs;
            }
        }
    }
}

// ---------------------------------------------------------------------------
// Launch
// ---------------------------------------------------------------------------
extern "C" void kernel_launch(void* const* d_in, const int* in_sizes, int n_in,
                              void* d_out, int out_size, void* d_ws, size_t ws_size,
                              hipStream_t stream) {
    const float* x        = (const float*)d_in[0];  // [4,1024,1024]
    const float* c_attn_w = (const float*)d_in[1];  // [1024,3072]
    const float* c_attn_b = (const float*)d_in[2];  // [3072]
    const float* c_proj_w = (const float*)d_in[3];  // [1024,1024]
    const float* c_proj_b = (const float*)d_in[4];  // [1024]
    float* out = (float*)d_out;

    char* w = (char*)d_ws;
    const size_t MB8 = (size_t)4096 * 1024 * 2;     // 8 MiB (bf16 4096x1024)
    bf16* xb  = (bf16*)w;                       w += MB8;
    bf16* wta = (bf16*)w;                       w += (size_t)3072 * 1024 * 2;
    bf16* wtp = (bf16*)w;                       w += (size_t)1024 * 1024 * 2;
    bf16* qb  = (bf16*)w;                       w += MB8;
    bf16* kb  = (bf16*)w;                       w += MB8;
    bf16* vtb = (bf16*)w;                       w += MB8;
    bf16* ab  = (bf16*)w;                       w += MB8;

    // 1) precision conversion / weight transposition
    cvt_f32_bf16<<<(4096 * 1024 + 255) / 256, 256, 0, stream>>>(x, xb, 4096 * 1024);
    transpose_cvt<<<(1024 * 3072 + 255) / 256, 256, 0, stream>>>(c_attn_w, wta, 1024, 3072);
    transpose_cvt<<<(1024 * 1024 + 255) / 256, 256, 0, stream>>>(c_proj_w, wtp, 1024, 1024);

    // 2) qkv projection (async-LDS staged B, double buffered, K step 64)
    gemm_qkv<<<dim3(32, 24), 256, 0, stream>>>(xb, wta, c_attn_b, qb, kb, vtb);

    // 3) causal attention (one wave per (b,h,16-row block))
    attn_kernel<<<4096, 32, 0, stream>>>(qb, kb, vtb, ab);

    // 4) output projection
    gemm_proj<<<dim3(32, 8), 256, 0, stream>>>(ab, wtp, c_proj_b, out);
}
